// EcocGPT2_7636451852628
// MI455X (gfx1250) — matrix operand order repaired and checked by reading, not compiled, not run on hardware
//
#include <hip/hip_runtime.h>
#include <hip/hip_bf16.h>

typedef __attribute__((ext_vector_type(16))) __bf16 v16bf;
typedef __attribute__((ext_vector_type(8)))  __bf16 v8bf;
typedef __attribute__((ext_vector_type(2)))  __bf16 v2bf;
typedef __attribute__((ext_vector_type(8)))  float  v8f;

#define BB 4
#define TT 512
#define CC 768
#define HH 12
#define HSZ 64
#define LLN 12
#define NBITS 16
#define RTILES 4   // row tiles per wave in k_gemm

// ---------------- embedding: x = tok_emb[idx] + pos_emb ----------------
__global__ void k_embed(const int* __restrict__ idx, const float* __restrict__ tok,
                        const float* __restrict__ pos, float* __restrict__ x) {
  int row = blockIdx.x;               // b*T + t
  int t   = row % TT;
  int id  = idx[row];
  const float4* te = (const float4*)(tok + (long)id * CC);
  const float4* pe = (const float4*)(pos + (long)t  * CC);
  float4* xr = (float4*)(x + (long)row * CC);
  for (int c = threadIdx.x; c < CC / 4; c += blockDim.x) {
    float4 a = te[c], p = pe[c];
    xr[c] = make_float4(a.x + p.x, a.y + p.y, a.z + p.z, a.w + p.w);
  }
}

// ---------------- layernorm: out_bf16 = LN(x)*g + b ----------------
__global__ void k_ln(const float* __restrict__ x, const float* __restrict__ g,
                     const float* __restrict__ bta, __bf16* __restrict__ out, int C) {
  int row = blockIdx.x;
  int tid  = threadIdx.x;
  int lane = tid & 31, wv = tid >> 5;
  const float4* xr4 = (const float4*)(x + (long)row * C);
  float s = 0.f, ss = 0.f;
  for (int c = tid; c < C / 4; c += blockDim.x) {
    float4 v = xr4[c];
    s  += v.x + v.y + v.z + v.w;
    ss += v.x * v.x + v.y * v.y + v.z * v.z + v.w * v.w;
  }
  // wave-level reduce, then one LDS combine
  for (int m = 1; m < 32; m <<= 1) { s += __shfl_xor(s, m, 32); ss += __shfl_xor(ss, m, 32); }
  __shared__ float rs[8], rss[8];
  if (lane == 0) { rs[wv] = s; rss[wv] = ss; }
  __syncthreads();
  float st = 0.f, sst = 0.f;
#pragma unroll
  for (int i = 0; i < 8; i++) { st += rs[i]; sst += rss[i]; }
  float mean = st / C;
  float var  = sst / C - mean * mean;
  float inv  = rsqrtf(var + 1e-5f);
  const float4* g4 = (const float4*)g;
  const float4* b4 = (const float4*)bta;
  for (int c = tid; c < C / 4; c += blockDim.x) {
    float4 v = xr4[c], gg = g4[c], bb = b4[c];
    __bf16* o = out + (long)row * C + c * 4;
    o[0] = (__bf16)((v.x - mean) * inv * gg.x + bb.x);
    o[1] = (__bf16)((v.y - mean) * inv * gg.y + bb.y);
    o[2] = (__bf16)((v.z - mean) * inv * gg.z + bb.z);
    o[3] = (__bf16)((v.w - mean) * inv * gg.w + bb.w);
  }
}

// ---------------- generic bf16-WMMA GEMM ----------------
// out[M,N] = A_bf16[M,K] * W_f32 (+bias)(+gelu)(+resid); weight address:
//   W[k*wsK + (n/64)*wsH + (n%64)] -- covers row-major [K,N] and QKV [H,C,HS].
// Block: 8 waves, each owning RTILES 16-row tiles (block = 512 rows x 16 cols);
// 32x16 weight tile staged transposed in LDS: one (n, k-pair) per thread.
__global__ void k_gemm(const __bf16* __restrict__ A, int K,
                       const float* __restrict__ W, long wsK, long wsH,
                       const float* __restrict__ bias,
                       const float* resid, float* outF, __bf16* outB, int N,
                       int qkvT, int act) {
  __shared__ __bf16 sBt[16 * 32];     // transposed weight tile: [n][k]
  int lane = threadIdx.x & 31;
  int wv   = threadIdx.x >> 5;
  int rbase = blockIdx.x * (128 * RTILES) + wv * 16;
  int col0  = blockIdx.y * 16;
  int half = lane >> 4;
  int l16  = lane & 15;

  // staging assignment: thread -> (nn, k-pair); all k-invariant math hoisted
  int nn = threadIdx.x & 15;
  int kp = threadIdx.x >> 4;          // 0..15 -> k = 2*kp, 2*kp+1
  {
  }
  int nstage = col0 + nn;
  const float* wptr = W + (long)(2 * kp) * wsK + (long)(nstage >> 6) * wsH + (nstage & 63);
  v2bf* sdst = (v2bf*)&sBt[nn * 32 + 2 * kp];
  const long wstep = 32 * wsK;

  // A row pointers hoisted out of the K loop
  const __bf16* ar[RTILES];
#pragma unroll
  for (int t = 0; t < RTILES; t++)
    ar[t] = A + (long)(rbase + t * 128 + l16) * K + half * 16;
  const v16bf* fbp = (const v16bf*)&sBt[l16 * 32 + half * 16];

  v8f acc[RTILES] = {{}, {}, {}, {}};

  for (int k0 = 0; k0 < K; k0 += 32) {
    // stage one bf16 pair per thread (fp32 weight stream -> bf16 LDS, transposed)
    float w0 = wptr[0];
    float w1 = wptr[wsK];
    v2bf pk = { (__bf16)w0, (__bf16)w1 };
    *sdst = pk;
    if (k0 + 32 < K) __builtin_prefetch(wptr + wstep, 0, 1);
    wptr += wstep;
    __syncthreads();

    v16bf fb = *fbp;                  // B fragment: contiguous 32B (k = half*16 + e)
#pragma unroll
    for (int t = 0; t < RTILES; t++) {
      v8bf lo = *(const v8bf*)(ar[t] + k0);         // k = e   + 16*half
      v8bf hi = *(const v8bf*)(ar[t] + k0 + 8);     // k = e+8 + 16*half
      v16bf fa = __builtin_shufflevector(lo, hi, 0, 1, 2, 3, 4, 5, 6, 7,
                                         8, 9, 10, 11, 12, 13, 14, 15);
      acc[t] = __builtin_amdgcn_wmma_f32_16x16x32_bf16(false, fa, false, fb,
                                                       (short)0, acc[t], false, false);
    }
    __syncthreads();
  }

  int n = col0 + l16;
  float bv = bias ? bias[n] : 0.f;
#pragma unroll
  for (int t = 0; t < RTILES; t++) {
#pragma unroll
    for (int v = 0; v < 8; v++) {
      int row = rbase + t * 128 + v + half * 8;     // C/D layout: VGPR v -> M=v(+8)
      float val = acc[t][v] + bv;
      if (act == 1) val = 0.5f * val * (1.f + erff(val * 0.70710678f));
      if (resid) val += resid[(long)row * N + n];
      if (qkvT > 0) {                               // q/k/v as [B*H, T, HS] bf16
        int b = row / qkvT, tt = row % qkvT;
        outB[(((long)(b * HH + (n >> 6)) * qkvT + tt) * HSZ) + (n & 63)] = (__bf16)val;
      } else if (outB) {
        outB[(long)row * N + n] = (__bf16)val;
      } else {
        outF[(long)row * N + n] = val;
      }
    }
  }
}

// ---------------- flash attention ----------------
// Block = 8 waves sharing one (b,h), covering 128 rows; K and V-transposed tiles
// staged cooperatively in LDS per 32-column block; all WMMA operands are
// contiguous vector loads, A and B both in k = 16*half + e order.
__global__ void k_attn(const __bf16* __restrict__ q, const __bf16* __restrict__ kmat,
                       const __bf16* __restrict__ vmat, __bf16* __restrict__ out) {
  __shared__ __bf16 sK[32 * 64];      // [s_rel][d]
  __shared__ __bf16 sVt[64 * 32];     // [d][s_rel]
  __shared__ __bf16 sP[8][16 * 32];   // per-wave P tile (transpose staging)
  int bh = blockIdx.x;
  int b = bh / HH, h = bh % HH;
  int r0 = blockIdx.y * 128;
  int wv   = threadIdx.x >> 5;
  int lane = threadIdx.x & 31;
  int half = lane >> 4, l16 = lane & 15;
  int t0 = r0 + wv * 16;

  // Q fragments over HS=64 (two K=32 chunks): vector loads + concat
  const __bf16* qrow = q + ((long)bh * TT + t0 + l16) * HSZ;
  v8bf qlo0 = *(const v8bf*)(qrow + half * 16);
  v8bf qhi0 = *(const v8bf*)(qrow + 8 + half * 16);
  v8bf qlo1 = *(const v8bf*)(qrow + 32 + half * 16);
  v8bf qhi1 = *(const v8bf*)(qrow + 40 + half * 16);
  v16bf fq0 = __builtin_shufflevector(qlo0, qhi0, 0,1,2,3,4,5,6,7,8,9,10,11,12,13,14,15);
  v16bf fq1 = __builtin_shufflevector(qlo1, qhi1, 0,1,2,3,4,5,6,7,8,9,10,11,12,13,14,15);

  float runm[8], runl[8];
  v8f o[4] = {{}, {}, {}, {}};
#pragma unroll
  for (int v = 0; v < 8; v++) { runm[v] = -1e30f; runl[v] = 0.f; }

  // staging assignments (k-invariant)
  const unsigned* gk  = (const unsigned*)(kmat + (long)bh * TT * HSZ);
  const unsigned* gv  = (const unsigned*)(vmat + (long)bh * TT * HSZ);
  unsigned* sk32 = (unsigned*)sK;

  int sendblk = r0 + 128;             // block-uniform loop bound
  for (int s0 = 0; s0 < sendblk; s0 += 32) {
    // ---- cooperative staging ----
    int base32 = s0 * 32;             // dword offset of this 32x64-bf16 tile
    for (int i = threadIdx.x; i < 1024; i += 256) sk32[i] = gk[base32 + i];
#pragma unroll
    for (int i = 0; i < 4; i++) {     // V transpose: 1024 dword pairs / 256 thr
      int e  = threadIdx.x + i * 256;
      int s  = e >> 5;                // 0..31
      int dp = (e & 31) * 2;          // even d
      unsigned u = gv[base32 + e];
      sVt[dp * 32 + s]       = __builtin_bit_cast(__bf16, (unsigned short)(u & 0xffffu));
      sVt[(dp + 1) * 32 + s] = __builtin_bit_cast(__bf16, (unsigned short)(u >> 16));
    }
    __syncthreads();

    if (s0 < t0 + 16) {               // causal: this wave still has live columns
      // S = Q K^T, two 16-col subtiles
      v8f sc[2] = {{}, {}};
#pragma unroll
      for (int sub = 0; sub < 2; sub++) {
        const __bf16* kr = &sK[(sub * 16 + l16) * 64];
        v16bf fb0 = *(const v16bf*)(kr + half * 16);        // d = 0..31
        v16bf fb1 = *(const v16bf*)(kr + 32 + half * 16);   // d = 32..63
        sc[sub] = __builtin_amdgcn_wmma_f32_16x16x32_bf16(false, fq0, false, fb0,
                                                          (short)0, sc[sub], false, false);
        sc[sub] = __builtin_amdgcn_wmma_f32_16x16x32_bf16(false, fq1, false, fb1,
                                                          (short)0, sc[sub], false, false);
      }
      // online softmax (row stats via shfl_xor within 16-lane halves)
      float pv0[8], pv1[8], mt[8];
#pragma unroll
      for (int v = 0; v < 8; v++) {
        int row = t0 + v + half * 8;
        float a0 = sc[0][v] * 0.125f; if (s0 + l16 > row)      a0 = -1e30f;
        float a1 = sc[1][v] * 0.125f; if (s0 + 16 + l16 > row) a1 = -1e30f;
        pv0[v] = a0; pv1[v] = a1;
        mt[v] = fmaxf(a0, a1);
      }
#pragma unroll
      for (int v = 0; v < 8; v++)
        for (int m = 1; m < 16; m <<= 1) mt[v] = fmaxf(mt[v], __shfl_xor(mt[v], m, 32));
#pragma unroll
      for (int v = 0; v < 8; v++) {
        float nm  = fmaxf(runm[v], mt[v]);
        float scl = __expf(runm[v] - nm);
        runm[v] = nm;
        o[0][v] *= scl; o[1][v] *= scl; o[2][v] *= scl; o[3][v] *= scl;
        float p0 = __expf(pv0[v] - nm);
        float p1 = __expf(pv1[v] - nm);
        float ps = p0 + p1;
        for (int m = 1; m < 16; m <<= 1) ps += __shfl_xor(ps, m, 32);
        runl[v] = runl[v] * scl + ps;
        sP[wv][(v + half * 8) * 32 + l16]      = (__bf16)p0;
        sP[wv][(v + half * 8) * 32 + 16 + l16] = (__bf16)p1;
      }
      // reload P as A fragment (same-wave LDS RAW: DS ops complete in order)
      v8bf plo = *(const v8bf*)&sP[wv][l16 * 32 + half * 16];
      v8bf phi = *(const v8bf*)&sP[wv][l16 * 32 + 8 + half * 16];
      v16bf fp = __builtin_shufflevector(plo, phi, 0,1,2,3,4,5,6,7,8,9,10,11,12,13,14,15);
      // O += P V over 4 HS column tiles; Vt gives contiguous B fragments
#pragma unroll
      for (int nt = 0; nt < 4; nt++) {
        v16bf fv = *(const v16bf*)&sVt[(nt * 16 + l16) * 32 + half * 16];
        o[nt] = __builtin_amdgcn_wmma_f32_16x16x32_bf16(false, fp, false, fv,
                                                        (short)0, o[nt], false, false);
      }
    }
    __syncthreads();
  }
  // normalize and write concat-head output [B*T, C] bf16
#pragma unroll
  for (int v = 0; v < 8; v++) {
    float inv = 1.f / runl[v];
    int row = t0 + v + half * 8;
    long base = ((long)(b * TT + row)) * CC + h * HSZ;
#pragma unroll
    for (int nt = 0; nt < 4; nt++)
      out[base + nt * 16 + l16] = (__bf16)(o[nt][v] * inv);
  }
}

// ---------------- host orchestration ----------------
extern "C" void kernel_launch(void* const* d_in, const int* in_sizes, int n_in,
                              void* d_out, int out_size, void* d_ws, size_t ws_size,
                              hipStream_t stream) {
  (void)in_sizes; (void)n_in; (void)out_size; (void)ws_size;
  const int*   idx  = (const int*)  d_in[0];
  const float* tok  = (const float*)d_in[1];
  const float* pos  = (const float*)d_in[2];
  const float* Wq   = (const float*)d_in[3];
  const float* Wk   = (const float*)d_in[4];
  const float* Wv   = (const float*)d_in[5];
  const float* Wo   = (const float*)d_in[6];
  const float* bo   = (const float*)d_in[7];
  const float* W1   = (const float*)d_in[8];
  const float* b1   = (const float*)d_in[9];
  const float* W2   = (const float*)d_in[10];
  const float* b2   = (const float*)d_in[11];
  const float* ln1g = (const float*)d_in[12];
  const float* ln1b = (const float*)d_in[13];
  const float* ln2g = (const float*)d_in[14];
  const float* ln2b = (const float*)d_in[15];
  const float* lnfg = (const float*)d_in[16];
  const float* lnfb = (const float*)d_in[17];
  const float* Wh   = (const float*)d_in[18];
  const float* bhp  = (const float*)d_in[19];

  const long M = (long)BB * TT;   // 2048 rows
  char* ws = (char*)d_ws;
  float*  x  = (float*) ws; ws += M * CC * 4;     // fp32 residual stream
  __bf16* xn = (__bf16*)ws; ws += M * CC * 2;     // LN output (bf16)
  __bf16* qb = (__bf16*)ws; ws += M * CC * 2;     // q  [B*H,T,HS]
  __bf16* kb = (__bf16*)ws; ws += M * CC * 2;     // k
  __bf16* vb = (__bf16*)ws; ws += M * CC * 2;     // v
  __bf16* ab = (__bf16*)ws; ws += M * CC * 2;     // attn concat-head out
  __bf16* hb = (__bf16*)ws; ws += M * 4 * CC * 2; // MLP hidden

  const int GX = (int)(M / (128 * RTILES));       // 4 row-blocks

  k_embed<<<(int)M, 256, 0, stream>>>(idx, tok, pos, x);

  for (int l = 0; l < LLN; l++) {
    long wqkv = (long)l * HH * CC * HSZ;
    k_ln<<<(int)M, 256, 0, stream>>>(x, ln1g + l * CC, ln1b + l * CC, xn, CC);
    k_gemm<<<dim3(GX, 48), 256, 0, stream>>>(xn, CC, Wq + wqkv, HSZ, (long)CC * HSZ,
                                             nullptr, nullptr, nullptr, qb, CC, TT, 0);
    k_gemm<<<dim3(GX, 48), 256, 0, stream>>>(xn, CC, Wk + wqkv, HSZ, (long)CC * HSZ,
                                             nullptr, nullptr, nullptr, kb, CC, TT, 0);
    k_gemm<<<dim3(GX, 48), 256, 0, stream>>>(xn, CC, Wv + wqkv, HSZ, (long)CC * HSZ,
                                             nullptr, nullptr, nullptr, vb, CC, TT, 0);
    k_attn<<<dim3(BB * HH, TT / 128), 256, 0, stream>>>(qb, kb, vb, ab);
    k_gemm<<<dim3(GX, 48), 256, 0, stream>>>(ab, CC, Wo + (long)l * CC * CC, CC, HSZ,
                                             bo + l * CC, x, x, nullptr, CC, 0, 0);
    k_ln<<<(int)M, 256, 0, stream>>>(x, ln2g + l * CC, ln2b + l * CC, xn, CC);
    k_gemm<<<dim3(GX, 192), 256, 0, stream>>>(xn, CC, W1 + (long)l * CC * 4 * CC,
                                              4 * CC, HSZ, b1 + (long)l * 4 * CC,
                                              nullptr, nullptr, hb, 4 * CC, 0, 1);
    k_gemm<<<dim3(GX, 48), 256, 0, stream>>>(hb, 4 * CC, W2 + (long)l * 4 * CC * CC,
                                             CC, HSZ, b2 + l * CC, x, x, nullptr, CC, 0, 0);
  }

  k_ln<<<(int)M, 256, 0, stream>>>(x, lnfg, lnfb, xn, CC);
  k_gemm<<<dim3(GX, 1), 256, 0, stream>>>(xn, CC, Wh, NBITS, HSZ, bhp,
                                          nullptr, (float*)d_out, nullptr, NBITS, 0, 0);
}
